// DemoAttentionModel_77575699300507
// MI455X (gfx1250) — matrix operand order repaired and checked by reading, not compiled, not run on hardware
//
#include <hip/hip_runtime.h>
#include <hip/hip_bf16.h>
#include <math.h>

// ---------------------------------------------------------------------------
// Transformer encoder block for MI455X (gfx1250, wave32, WMMA).
// All matmuls via v_wmma_f32_16x16x32_f16 (f16 in, f32 accum).
// Block-shared tiles (GEMM B tile, attention K/V tiles) are staged with
// GLOBAL_LOAD_ASYNC_TO_LDS_B128 (ASYNCcnt) and double-buffered against WMMA.
// Fragments use contiguous k order (k = 16*(lane/16) + s) for both A and B.
// ---------------------------------------------------------------------------

#define D_MODEL   512
#define NUM_HEADS 8
#define HEAD_DIM  64
#define SEQ_LEN   4096
#define BATCH     2
#define D_FF      2048
#define ROWS      (BATCH * SEQ_LEN)   // 8192

typedef __attribute__((ext_vector_type(16))) _Float16 v16h;
typedef __attribute__((ext_vector_type(8)))  _Float16 v8h;
typedef __attribute__((ext_vector_type(8)))  float    v8f;

typedef __attribute__((address_space(3))) char lds_char_t;

static __device__ __forceinline__ v16h make_v16h(v8h lo, v8h hi) {
  union { v16h v; v8h h[2]; } u;
  u.h[0] = lo; u.h[1] = hi;
  return u.v;
}

// A-fragment load from a row-major row (two contiguous 8-half runs per lane).
static __device__ __forceinline__ v16h load_a_frag(const _Float16* __restrict__ row,
                                                   int k0, int lane) {
  const int off = (lane < 16) ? 0 : 16;
  v8h lo = *(const v8h*)(row + k0 + off);
  v8h hi = *(const v8h*)(row + k0 + off + 8);
  return make_v16h(lo, hi);
}

static __device__ __forceinline__ v8f wmma_f16(v16h a, v16h b, v8f c) {
  return __builtin_amdgcn_wmma_f32_16x16x32_f16(false, a, false, b, (short)0, c,
                                                false, false);
}

// Async 16-byte global -> LDS copy (per-lane addresses, tracked by ASYNCcnt).
static __device__ __forceinline__ void async_copy_b128(const _Float16* gsrc,
                                                       _Float16* ldst) {
  unsigned loff = (unsigned)(size_t)(lds_char_t*)(void*)ldst;  // raw LDS byte offset
  asm volatile("global_load_async_to_lds_b128 %0, %1, off"
               :: "v"(loff), "v"(gsrc)
               : "memory");
}

static __device__ __forceinline__ void wait_async0() {
  asm volatile("s_wait_asynccnt 0x0" ::: "memory");
}

// ---------------------------------------------------------------------------
// Weight transpose + cast: W[K][N] f32 -> Wt[N][K] f16 (B-matrix friendly).
// ---------------------------------------------------------------------------
__global__ __launch_bounds__(256) void transpose_cast_kernel(
    const float* __restrict__ W, _Float16* __restrict__ Wt, int K, int N) {
  const int n = blockIdx.x * 32 + (threadIdx.x & 31);
  const int k = blockIdx.y * 8 + (threadIdx.x >> 5);
  if (n < N && k < K) Wt[(size_t)n * K + k] = (_Float16)W[(size_t)k * N + n];
}

// ---------------------------------------------------------------------------
// LayerNorm over D_MODEL=512 and cast to f16. One block per row.
// ---------------------------------------------------------------------------
__global__ __launch_bounds__(256) void layernorm_cast_kernel(
    const float* __restrict__ x, const float* __restrict__ g,
    const float* __restrict__ b, _Float16* __restrict__ out) {
  __shared__ float red[256];
  const int row = blockIdx.x;
  const int t = threadIdx.x;
  const float* xr = x + (size_t)row * D_MODEL;
  const float v0 = xr[t];
  const float v1 = xr[t + 256];

  red[t] = v0 + v1;
  __syncthreads();
#pragma unroll
  for (int o = 128; o > 0; o >>= 1) {
    if (t < o) red[t] += red[t + o];
    __syncthreads();
  }
  const float mean = red[0] * (1.0f / (float)D_MODEL);
  __syncthreads();

  const float d0 = v0 - mean, d1 = v1 - mean;
  red[t] = d0 * d0 + d1 * d1;
  __syncthreads();
#pragma unroll
  for (int o = 128; o > 0; o >>= 1) {
    if (t < o) red[t] += red[t + o];
    __syncthreads();
  }
  const float var = red[0] * (1.0f / (float)D_MODEL);
  const float rstd = rsqrtf(var + 1e-5f);

  _Float16* orow = out + (size_t)row * D_MODEL;
  orow[t]       = (_Float16)(d0 * rstd * g[t] + b[t]);
  orow[t + 256] = (_Float16)(d1 * rstd * g[t + 256] + b[t + 256]);
}

// ---------------------------------------------------------------------------
// WMMA GEMM: C[M,N] = A[M,K] @ Bt[N,K]^T + bias, with fused epilogues.
// Block = 256 threads = 8 waves; block tile 128(M) x 64(N); wave tile 16x64.
// B tile (shared by all waves) staged via async LDS copies, double buffered.
// ---------------------------------------------------------------------------
enum { MODE_QK = 0, MODE_VT = 1, MODE_RESID_F32 = 2, MODE_GELU_F16 = 3 };

template <int MODE>
__global__ __launch_bounds__(256) void gemm_wmma_kernel(
    const _Float16* __restrict__ A, const _Float16* __restrict__ Bt,
    const float* __restrict__ bias, const float* __restrict__ resid,
    void* __restrict__ outp, int K, int N) {
  __shared__ __align__(16) _Float16 tileB[2][64][40];  // 64 N-rows x 32 K-halves, padded

  const int lane = threadIdx.x & 31;
  const int wave = threadIdx.x >> 5;
  const int mBase = blockIdx.y * 128 + wave * 16;
  const int nBase = blockIdx.x * 64;

  // cooperative async stage of one 64x32 B tile (256 x 16B chunks)
  auto prefetchB = [&](int buf, int k0) {
    const int row = threadIdx.x >> 2;
    const int c8 = (threadIdx.x & 3) * 8;
    async_copy_b128(Bt + (size_t)(nBase + row) * K + k0 + c8, &tileB[buf][row][c8]);
  };

  v8f acc[4];
  const v8f z = {};
#pragma unroll
  for (int i = 0; i < 4; ++i) acc[i] = z;

  const _Float16* Arow = A + (size_t)(mBase + (lane & 15)) * K;
  const int boff = (lane < 16) ? 0 : 16;

  prefetchB(0, 0);
  wait_async0();
  __syncthreads();

  int buf = 0;
  for (int k0 = 0; k0 < K; k0 += 32) {
    const bool more = (k0 + 32) < K;
    if (more) prefetchB(buf ^ 1, k0 + 32);           // overlap next B tile
    if (more) __builtin_prefetch(Arow + k0 + 32, 0, 1);  // global_prefetch_b8

    const v16h a = load_a_frag(Arow, k0, lane);
#pragma unroll
    for (int nt = 0; nt < 4; ++nt) {
      const v16h b = *(const v16h*)&tileB[buf][nt * 16 + (lane & 15)][boff];
      acc[nt] = wmma_f16(a, b, acc[nt]);
    }

    if (more) wait_async0();                          // next tile landed (this wave)
    __syncthreads();                                  // all waves done reading buf
    buf ^= 1;
  }

#pragma unroll
  for (int nt = 0; nt < 4; ++nt) {
    const int col = nBase + nt * 16 + (lane & 15);
    const float bcol = bias[col];
#pragma unroll
    for (int r = 0; r < 8; ++r) {
      const int mloc = (lane < 16) ? r : (8 + r);
      const int row = mBase + mloc;
      const float val = acc[nt][r] + bcol;
      if (MODE == MODE_QK) {
        const int bb = row >> 12, s = row & (SEQ_LEN - 1);
        const int hh = col >> 6, d = col & 63;
        ((_Float16*)outp)[((size_t)(bb * NUM_HEADS + hh) * SEQ_LEN + s) * HEAD_DIM + d] =
            (_Float16)val;
      } else if (MODE == MODE_VT) {
        const int bb = row >> 12, s = row & (SEQ_LEN - 1);
        const int hh = col >> 6, d = col & 63;
        ((_Float16*)outp)[((size_t)(bb * NUM_HEADS + hh) * HEAD_DIM + d) * SEQ_LEN + s] =
            (_Float16)val;
      } else if (MODE == MODE_RESID_F32) {
        ((float*)outp)[(size_t)row * N + col] = val + resid[(size_t)row * N + col];
      } else {  // MODE_GELU_F16, exact gelu
        const float gl = 0.5f * val * (1.0f + erff(val * 0.70710678f));
        ((_Float16*)outp)[(size_t)row * N + col] = (_Float16)gl;
      }
    }
  }
}

// ---------------------------------------------------------------------------
// Flash attention. grid = (S/128, B*H). Block 256 = 8 waves, wave owns 16
// query rows. K/V tiles (64 keys x 64 dh) staged via async LDS, double
// buffered; online softmax; P staged via wave-private LDS tile and reloaded
// as A-fragments for the PV WMMAs. Output scattered to attn[B*S, D_MODEL].
// ---------------------------------------------------------------------------
__global__ __launch_bounds__(256) void flash_attn_kernel(
    const _Float16* __restrict__ q, const _Float16* __restrict__ k,
    const _Float16* __restrict__ vt, _Float16* __restrict__ attn) {
  __shared__ __align__(16) _Float16 ktile[2][64][72];  // [key][dh], padded
  __shared__ __align__(16) _Float16 vtile[2][64][72];  // [dh][key], padded
  __shared__ __align__(16) _Float16 plds[8][16][72];   // per-wave P tile

  const int lane = threadIdx.x & 31;
  const int wave = threadIdx.x >> 5;
  const int bh = blockIdx.y;              // b*NUM_HEADS + h
  const int bb = bh >> 3, hh = bh & 7;
  const int q0 = blockIdx.x * 128 + wave * 16;

  const _Float16* qbase = q + ((size_t)bh * SEQ_LEN + q0) * HEAD_DIM;
  const _Float16* kbase = k + (size_t)bh * SEQ_LEN * HEAD_DIM;
  const _Float16* vbase = vt + (size_t)bh * HEAD_DIM * SEQ_LEN;

  // cooperative async stage of 64x64 K tile + 64x64 V^T tile (2x512 chunks)
  auto prefetchKV = [&](int buf, int kt) {
#pragma unroll
    for (int i = 0; i < 2; ++i) {
      const int c = threadIdx.x + i * 256;
      const int row = c >> 3;
      const int c8 = (c & 7) * 8;
      async_copy_b128(kbase + (size_t)(kt + row) * HEAD_DIM + c8, &ktile[buf][row][c8]);
      async_copy_b128(vbase + (size_t)row * SEQ_LEN + kt + c8, &vtile[buf][row][c8]);
    }
  };

  prefetchKV(0, 0);

  const int mrow = lane & 15;
  const int boff = (lane < 16) ? 0 : 16;
  const _Float16* qrow = qbase + mrow * HEAD_DIM;
  const v16h aq0 = load_a_frag(qrow, 0, lane);
  const v16h aq1 = load_a_frag(qrow, 32, lane);

  v8f o[4];
  const v8f z = {};
#pragma unroll
  for (int i = 0; i < 4; ++i) o[i] = z;

  float mrun[8], lrun[8];
#pragma unroll
  for (int r = 0; r < 8; ++r) { mrun[r] = -INFINITY; lrun[r] = 0.0f; }

  const float scale = 0.125f;  // 1/sqrt(64)

  wait_async0();
  __syncthreads();

  int buf = 0;
  for (int kt = 0; kt < SEQ_LEN; kt += 64) {
    const bool more = (kt + 64) < SEQ_LEN;
    if (more) prefetchKV(buf ^ 1, kt + 64);

    // ---- scores S = (Q @ K^T) * scale : 16 x 64 tile, 4 N-subtiles ----
    v8f s[4];
#pragma unroll
    for (int i = 0; i < 4; ++i) s[i] = z;
#pragma unroll
    for (int nt = 0; nt < 4; ++nt) {
      const v16h b0 = *(const v16h*)&ktile[buf][nt * 16 + mrow][boff];
      s[nt] = wmma_f16(aq0, b0, s[nt]);
      const v16h b1 = *(const v16h*)&ktile[buf][nt * 16 + mrow][32 + boff];
      s[nt] = wmma_f16(aq1, b1, s[nt]);
    }

    // ---- online softmax (rows live in VGPR index r across a 16-lane half) --
    float tmax[8];
#pragma unroll
    for (int r = 0; r < 8; ++r) {
      float v = s[0][r] * scale;
      v = fmaxf(v, s[1][r] * scale);
      v = fmaxf(v, s[2][r] * scale);
      v = fmaxf(v, s[3][r] * scale);
#pragma unroll
      for (int off = 1; off < 16; off <<= 1)
        v = fmaxf(v, __shfl_xor(v, off, 32));
      tmax[r] = v;
    }

    float alpha[8];
#pragma unroll
    for (int r = 0; r < 8; ++r) {
      const float mn = fmaxf(mrun[r], tmax[r]);
      alpha[r] = __expf(mrun[r] - mn);
      mrun[r] = mn;
    }
#pragma unroll
    for (int nt = 0; nt < 4; ++nt)
#pragma unroll
      for (int r = 0; r < 8; ++r)
        s[nt][r] = __expf(s[nt][r] * scale - mrun[r]);

#pragma unroll
    for (int r = 0; r < 8; ++r) {
      float ts = s[0][r] + s[1][r] + s[2][r] + s[3][r];
#pragma unroll
      for (int off = 1; off < 16; off <<= 1) ts += __shfl_xor(ts, off, 32);
      lrun[r] = lrun[r] * alpha[r] + ts;
    }
#pragma unroll
    for (int nt = 0; nt < 4; ++nt)
#pragma unroll
      for (int r = 0; r < 8; ++r) o[nt][r] *= alpha[r];

    // ---- stage P (f16) into wave-private LDS tile ----
#pragma unroll
    for (int nt = 0; nt < 4; ++nt)
#pragma unroll
      for (int r = 0; r < 8; ++r) {
        const int ml = (lane < 16) ? r : (8 + r);
        plds[wave][ml][nt * 16 + mrow] = (_Float16)s[nt][r];
      }
    __builtin_amdgcn_wave_barrier();  // wave-private buffer; DS ops are in-order

    // ---- O += P @ V : A = P from LDS, B = V^T rows from LDS ----
#pragma unroll
    for (int nt = 0; nt < 4; ++nt) {
#pragma unroll
      for (int ks = 0; ks < 64; ks += 32) {
        const int aoff = (lane < 16) ? 0 : 16;
        const v8h lo = *(const v8h*)&plds[wave][mrow][ks + aoff];
        const v8h hi = *(const v8h*)&plds[wave][mrow][ks + aoff + 8];
        const v16h ap = make_v16h(lo, hi);
        const v16h bv = *(const v16h*)&vtile[buf][nt * 16 + mrow][ks + boff];
        o[nt] = wmma_f16(ap, bv, o[nt]);
      }
    }

    if (more) wait_async0();   // next K/V tile landed (this wave)
    __syncthreads();           // all waves done reading buf
    buf ^= 1;
  }

  // ---- finalize: O /= l, scatter heads back to [B*S, D_MODEL] ----
#pragma unroll
  for (int nt = 0; nt < 4; ++nt)
#pragma unroll
    for (int r = 0; r < 8; ++r) {
      const int ml = (lane < 16) ? r : (8 + r);
      const int srow = q0 + ml;
      attn[((size_t)bb * SEQ_LEN + srow) * D_MODEL + hh * HEAD_DIM + nt * 16 + mrow] =
          (_Float16)(o[nt][r] / lrun[r]);
    }
}

// ---------------------------------------------------------------------------
// Host-side orchestration.
// ---------------------------------------------------------------------------
extern "C" void kernel_launch(void* const* d_in, const int* in_sizes, int n_in,
                              void* d_out, int out_size, void* d_ws, size_t ws_size,
                              hipStream_t stream) {
  const float* x    = (const float*)d_in[0];
  const float* Wq   = (const float*)d_in[1];
  const float* bq   = (const float*)d_in[2];
  const float* Wk   = (const float*)d_in[3];
  const float* bk   = (const float*)d_in[4];
  const float* Wv   = (const float*)d_in[5];
  const float* bv   = (const float*)d_in[6];
  const float* Wo   = (const float*)d_in[7];
  const float* bo   = (const float*)d_in[8];
  const float* ln1g = (const float*)d_in[9];
  const float* ln1b = (const float*)d_in[10];
  const float* ln2g = (const float*)d_in[11];
  const float* ln2b = (const float*)d_in[12];
  const float* W1   = (const float*)d_in[13];
  const float* b1   = (const float*)d_in[14];
  const float* W2   = (const float*)d_in[15];
  const float* b2   = (const float*)d_in[16];

  char* ws = (char*)d_ws;
  size_t off = 0;
  _Float16* WqT = (_Float16*)(ws + off); off += (size_t)D_MODEL * D_MODEL * 2;
  _Float16* WkT = (_Float16*)(ws + off); off += (size_t)D_MODEL * D_MODEL * 2;
  _Float16* WvT = (_Float16*)(ws + off); off += (size_t)D_MODEL * D_MODEL * 2;
  _Float16* WoT = (_Float16*)(ws + off); off += (size_t)D_MODEL * D_MODEL * 2;
  _Float16* W1T = (_Float16*)(ws + off); off += (size_t)D_FF * D_MODEL * 2;
  _Float16* W2T = (_Float16*)(ws + off); off += (size_t)D_MODEL * D_FF * 2;
  _Float16* h16 = (_Float16*)(ws + off); off += (size_t)ROWS * D_MODEL * 2;   // ln1 out, reused for ln2 out
  float*    x1  = (float*)(ws + off);    off += (size_t)ROWS * D_MODEL * 4;   // residual 1
  // q/k/vt/attn region (4 x 8MB); reused afterwards as a1 (8192 x 2048 f16 = 32MB)
  _Float16* qb   = (_Float16*)(ws + off);
  _Float16* a1   = (_Float16*)(ws + off); off += (size_t)ROWS * D_MODEL * 2;
  _Float16* kb   = (_Float16*)(ws + off); off += (size_t)ROWS * D_MODEL * 2;
  _Float16* vtb  = (_Float16*)(ws + off); off += (size_t)ROWS * D_MODEL * 2;
  _Float16* attn = (_Float16*)(ws + off); off += (size_t)ROWS * D_MODEL * 2;
  (void)off; (void)ws_size; (void)in_sizes; (void)n_in; (void)out_size;

  dim3 blk(256);

  // 1) weight transpose+cast
  transpose_cast_kernel<<<dim3(D_MODEL / 32, D_MODEL / 8), blk, 0, stream>>>(Wq, WqT, D_MODEL, D_MODEL);
  transpose_cast_kernel<<<dim3(D_MODEL / 32, D_MODEL / 8), blk, 0, stream>>>(Wk, WkT, D_MODEL, D_MODEL);
  transpose_cast_kernel<<<dim3(D_MODEL / 32, D_MODEL / 8), blk, 0, stream>>>(Wv, WvT, D_MODEL, D_MODEL);
  transpose_cast_kernel<<<dim3(D_MODEL / 32, D_MODEL / 8), blk, 0, stream>>>(Wo, WoT, D_MODEL, D_MODEL);
  transpose_cast_kernel<<<dim3(D_FF / 32, D_MODEL / 8), blk, 0, stream>>>(W1, W1T, D_MODEL, D_FF);
  transpose_cast_kernel<<<dim3(D_MODEL / 32, D_FF / 8), blk, 0, stream>>>(W2, W2T, D_FF, D_MODEL);

  // 2) LN1 -> h16
  layernorm_cast_kernel<<<dim3(ROWS), blk, 0, stream>>>(x, ln1g, ln1b, h16);

  // 3) QKV projections (head scatter; V transposed)
  dim3 g512(D_MODEL / 64, ROWS / 128);
  gemm_wmma_kernel<MODE_QK><<<g512, blk, 0, stream>>>(h16, WqT, bq, nullptr, qb, D_MODEL, D_MODEL);
  gemm_wmma_kernel<MODE_QK><<<g512, blk, 0, stream>>>(h16, WkT, bk, nullptr, kb, D_MODEL, D_MODEL);
  gemm_wmma_kernel<MODE_VT><<<g512, blk, 0, stream>>>(h16, WvT, bv, nullptr, vtb, D_MODEL, D_MODEL);

  // 4) attention
  flash_attn_kernel<<<dim3(SEQ_LEN / 128, BATCH * NUM_HEADS), blk, 0, stream>>>(qb, kb, vtb, attn);

  // 5) output projection + residual -> x1 (f32)
  gemm_wmma_kernel<MODE_RESID_F32><<<g512, blk, 0, stream>>>(attn, WoT, bo, x, x1, D_MODEL, D_MODEL);

  // 6) LN2 -> h16 (reuse)
  layernorm_cast_kernel<<<dim3(ROWS), blk, 0, stream>>>(x1, ln2g, ln2b, h16);

  // 7) FFN1 + GELU -> a1 (f16)   (a1 reuses q/k/vt/attn region)
  gemm_wmma_kernel<MODE_GELU_F16><<<dim3(D_FF / 64, ROWS / 128), blk, 0, stream>>>(
      h16, W1T, b1, nullptr, a1, D_MODEL, D_FF);

  // 8) FFN2 + residual -> d_out (f32)
  gemm_wmma_kernel<MODE_RESID_F32><<<g512, blk, 0, stream>>>(
      a1, W2T, b2, x1, (float*)d_out, D_FF, D_MODEL);
}
